// NetVLAD_49581102465401
// MI455X (gfx1250) — matrix operand order repaired and checked by reading, not compiled, not run on hardware
//
#include <hip/hip_runtime.h>

// NetVLAD fused pipeline for MI455X (gfx1250, wave32, WMMA f16->f32).
// Shapes: N=32, D=128, P=1024 (=32x32), K=64.

typedef __attribute__((ext_vector_type(16))) _Float16 v16h;
typedef __attribute__((ext_vector_type(8)))  float    v8f;

#define NB 32
#define DD 128
#define PN 1024
#define KC 64
#define PITCH 136   // LDS row pitch in halves (padding vs bank conflicts)
#define LOG2E 1.4426950408889634f

// ---------------------------------------------------------------------------
// WMMA fragment gathers from LDS, per CDNA5 ISA VGPR layouts (CDNA5 ISA).
// A (16x32 f16): lanes 0-15 hold rows M=0..15; VGPR j holds K pair
//   {2j (j<4) | 2j+8 (j>=4)} + 8*(lane>>4).     (two contiguous 16B runs)
// B (32x16 f16): lanes 0-15 cols; VGPR j holds K pair
//   {2j (j<4) | 2j+8 (j>=4)} + 8*(lane>>4).     (same K order as A)
// C (16x16 f32): element (M = r + 8*(lane>>4), Ncol = lane&15) in VGPR r.
// ---------------------------------------------------------------------------
__device__ __forceinline__ v16h load_a_frag(const _Float16* lds, int row_base,
                                            int kbase, int lane) {
  const int m = lane & 15, h = lane >> 4;
  const _Float16* r = lds + (row_base + m) * PITCH + kbase + 8 * h;
  v16h a;
#pragma unroll
  for (int j = 0; j < 8; ++j) {
    const int k0 = (j < 4) ? (2 * j) : (2 * j + 8);
    a[2 * j]     = r[k0];
    a[2 * j + 1] = r[k0 + 1];
  }
  return a;
}

__device__ __forceinline__ v16h load_b_frag(const _Float16* lds, int col_base,
                                            int kbase, int lane) {
  const int c = lane & 15, h = lane >> 4;
  const _Float16* r = lds + (col_base + c) * PITCH + kbase + 8 * h;
  v16h b;
#pragma unroll
  for (int j = 0; j < 8; ++j) {
    const int k0 = (j < 4) ? (2 * j) : (2 * j + 8);
    b[2 * j]     = r[k0];
    b[2 * j + 1] = r[k0 + 1];
  }
  return b;
}

// Stage [pn pixels][128 d] tile into LDS (transposed) from global [d][P] f16.
// lgpu: log2(pn/2) (uints per d-row). 2x unrolled so both global loads are
// in flight before the dependent LDS stores (avoids per-element loadcnt wait).
__device__ __forceinline__ void stage_T(_Float16* dst, const _Float16* g,
                                        int pbase, int lgpu, int tid) {
  const int total = 128 << lgpu;     // multiple of 512
  const int mask  = (1 << lgpu) - 1;
  for (int idx = tid; idx < total; idx += 512) {
    const int j   = idx + 256;
    const int d0  = idx >> lgpu, pu0 = idx & mask;
    const int d1  = j   >> lgpu, pu1 = j   & mask;
    const unsigned int v0 = *(const unsigned int*)(g + (size_t)d0 * PN + pbase + 2 * pu0);
    const unsigned int v1 = *(const unsigned int*)(g + (size_t)d1 * PN + pbase + 2 * pu1);
    union { unsigned int u; _Float16 h[2]; } a, b; a.u = v0; b.u = v1;
    dst[(2 * pu0) * PITCH + d0]     = a.h[0];
    dst[(2 * pu0 + 1) * PITCH + d0] = a.h[1];
    dst[(2 * pu1) * PITCH + d1]     = b.h[0];
    dst[(2 * pu1 + 1) * PITCH + d1] = b.h[1];
  }
}

// Stage nrows x 128 halves straight (row-major both sides), gpitch = 1024 halves.
__device__ __forceinline__ void stage_rows(_Float16* dst, const _Float16* g,
                                           int nrows, int tid) {
  const int total = nrows << 6;      // nrows * 64 uints, multiple of 512
  for (int idx = tid; idx < total; idx += 512) {
    const int j  = idx + 256;
    const int r0 = idx >> 6, pu0 = idx & 63;
    const int r1 = j >> 6,   pu1 = j & 63;
    const unsigned int v0 = *(const unsigned int*)(g + (size_t)r0 * PN + 2 * pu0);
    const unsigned int v1 = *(const unsigned int*)(g + (size_t)r1 * PN + 2 * pu1);
    *(unsigned int*)(dst + r0 * PITCH + 2 * pu0) = v0;
    *(unsigned int*)(dst + r1 * PITCH + 2 * pu1) = v1;
  }
}

// ---------------------------------------------------------------------------
// K1: per-pixel L2 normalize over D; emit f16 xf in [N, D, P] layout.
// ---------------------------------------------------------------------------
__global__ __launch_bounds__(256) void k1_norm(const float* __restrict__ x,
                                               _Float16* __restrict__ xf16) {
  const int n = blockIdx.y;
  const int p = blockIdx.x * 256 + threadIdx.x;
  const float* xp = x + (size_t)n * DD * PN + p;
  float ss = 0.f;
#pragma unroll 8
  for (int d = 0; d < DD; ++d) { const float v = xp[d * PN]; ss = fmaf(v, v, ss); }
  const float rn = 1.0f / fmaxf(sqrtf(ss), 1e-12f);
  _Float16* op = xf16 + (size_t)n * DD * PN + p;
#pragma unroll 8
  for (int d = 0; d < DD; ++d) op[d * PN] = (_Float16)(xp[d * PN] * rn);
}

// ---------------------------------------------------------------------------
// K2: fused Gram / antiburst. Block = (n, 128-row panel), 8 waves, 1 row
// tile each. A-fragments (4 K-chunks) held in registers; stream 16 panels
// of 64 columns through LDS. inv_w[n,p] = (sum_q sigmoid(...))^(-ab_p).
// sigmoid((2g-2)*w + b) == rcp(1 + exp2(g*c1 + c0)),
//   c1 = -2w*log2e, c0 = (2w - b)*log2e   (fast v_exp/v_rcp path).
// ---------------------------------------------------------------------------
__global__ __launch_bounds__(256) void k2_antiburst(const _Float16* __restrict__ xf16,
                                                    const float* __restrict__ abp,
                                                    float* __restrict__ invw) {
  __shared__ __align__(16) _Float16 smA[128 * PITCH];  // 34816 B
  __shared__ __align__(16) _Float16 smB[64 * PITCH];   // 17408 B
  const int n = blockIdx.y, rp = blockIdx.x;
  const int tid = threadIdx.x, wave = tid >> 5, lane = tid & 31;
  const _Float16* g = xf16 + (size_t)n * DD * PN;
  const float abw = abp[0], abb = abp[1], abpw = abp[2];
  const float c1 = -2.0f * abw * LOG2E;
  const float c0 = (2.0f * abw - abb) * LOG2E;

  stage_T(smA, g, rp * 128, 6, tid);
  __syncthreads();

  v16h aa[4];
#pragma unroll
  for (int kc = 0; kc < 4; ++kc) aa[kc] = load_a_frag(smA, wave * 16, kc * 32, lane);

  float acc[8];
#pragma unroll
  for (int r = 0; r < 8; ++r) acc[r] = 0.f;

  for (int cp = 0; cp < 16; ++cp) {
    __syncthreads();
    stage_T(smB, g, cp * 64, 5, tid);
    __syncthreads();
#pragma unroll
    for (int ct = 0; ct < 4; ++ct) {
      v16h bb[4];
#pragma unroll
      for (int kc = 0; kc < 4; ++kc) bb[kc] = load_b_frag(smB, ct * 16, kc * 32, lane);
      v8f c = {0.f, 0.f, 0.f, 0.f, 0.f, 0.f, 0.f, 0.f};
#pragma unroll
      for (int kc = 0; kc < 4; ++kc)
        c = __builtin_amdgcn_wmma_f32_16x16x32_f16(false, aa[kc], false, bb[kc],
                                                   (short)0, c, false, false);
#pragma unroll
      for (int r = 0; r < 8; ++r) {
        const float e = __builtin_amdgcn_exp2f(fmaf(c[r], c1, c0));
        acc[r] += __builtin_amdgcn_rcpf(1.0f + e);
      }
    }
  }

  // Reduce each row over the 16 lanes holding its columns.
#pragma unroll
  for (int r = 0; r < 8; ++r) {
    float v = acc[r];
#pragma unroll
    for (int m = 1; m <= 8; m <<= 1) v += __shfl_xor(v, m, 32);
    if ((lane & 15) == 0) {
      const int row = (lane >> 4) * 8 + r;             // M = r + 8h
      const int p   = rp * 128 + wave * 16 + row;
      invw[n * PN + p] = __powf(v, -abpw);             // 1 / w_burst^p
    }
  }
}

// ---------------------------------------------------------------------------
// K3: logits = conv_w @ xf (WMMA), softmax over K=64, scale by inv_w, emit
// f16 a[N,K,P] + deterministic per-panel partial column sums.
// Block = (n, 64-pixel panel).
// ---------------------------------------------------------------------------
__global__ __launch_bounds__(256) void k3_assign(const _Float16* __restrict__ xf16,
                                                 const float* __restrict__ conv_w,
                                                 const float* __restrict__ invw,
                                                 _Float16* __restrict__ a_out,
                                                 float* __restrict__ asum_part) {
  __shared__ __align__(16) _Float16 smCW[64 * PITCH];  // conv_w f16 [k][d]
  __shared__ __align__(16) _Float16 smX[64 * PITCH];   // xf tile [p][d]
  __shared__ __align__(16) float    smL[64 * 68];      // logits -> a values
  const int n = blockIdx.y, pp = blockIdx.x;           // 16 panels of 64 pixels
  const int tid = threadIdx.x, wave = tid >> 5, lane = tid & 31;

  for (int idx = tid; idx < KC * DD; idx += 256)
    smCW[(idx >> 7) * PITCH + (idx & 127)] = (_Float16)conv_w[idx];
  stage_T(smX, xf16 + (size_t)n * DD * PN, pp * 64, 5, tid);
  __syncthreads();

  const int kt = wave >> 1;                 // k row tile 0..3
  v16h aa[4];
#pragma unroll
  for (int kc = 0; kc < 4; ++kc) aa[kc] = load_a_frag(smCW, kt * 16, kc * 32, lane);

#pragma unroll
  for (int cti = 0; cti < 2; ++cti) {
    const int ct = (wave & 1) * 2 + cti;    // col tile 0..3
    v16h bb[4];
#pragma unroll
    for (int kc = 0; kc < 4; ++kc) bb[kc] = load_b_frag(smX, ct * 16, kc * 32, lane);
    v8f c = {0.f, 0.f, 0.f, 0.f, 0.f, 0.f, 0.f, 0.f};
#pragma unroll
    for (int kc = 0; kc < 4; ++kc)
      c = __builtin_amdgcn_wmma_f32_16x16x32_f16(false, aa[kc], false, bb[kc],
                                                 (short)0, c, false, false);
    const int colg = ct * 16 + (lane & 15);
#pragma unroll
    for (int r = 0; r < 8; ++r)
      smL[(kt * 16 + r + 8 * (lane >> 4)) * 68 + colg] = c[r];
  }
  __syncthreads();

  // Softmax over K=64: 4 threads per pixel, 16 clusters each.
  const int p_loc = tid >> 2, part = tid & 3;
  float mx = -1e30f;
#pragma unroll
  for (int i = 0; i < 16; ++i) mx = fmaxf(mx, smL[(part * 16 + i) * 68 + p_loc]);
  mx = fmaxf(mx, __shfl_xor(mx, 1, 32));
  mx = fmaxf(mx, __shfl_xor(mx, 2, 32));
  float e[16], s = 0.f;
#pragma unroll
  for (int i = 0; i < 16; ++i) {
    e[i] = __expf(smL[(part * 16 + i) * 68 + p_loc] - mx);
    s += e[i];
  }
  s += __shfl_xor(s, 1, 32);
  s += __shfl_xor(s, 2, 32);
  const int pg = pp * 64 + p_loc;
  const float scale = invw[n * PN + pg] / s;
#pragma unroll
  for (int i = 0; i < 16; ++i) {
    const int k = part * 16 + i;
    const float av = e[i] * scale;
    a_out[((size_t)n * KC + k) * PN + pg] = (_Float16)av;
    smL[k * 68 + p_loc] = av;              // in-place: cell owned by this thread
  }
  __syncthreads();

  // Deterministic column-sum partials: sum_p a[k,p] over this 64-pixel panel.
  const int krow = tid >> 2, quarter = tid & 3;
  float rs = 0.f;
#pragma unroll
  for (int i = 0; i < 16; ++i) rs += smL[krow * 68 + quarter * 16 + i];
  rs += __shfl_xor(rs, 1, 32);
  rs += __shfl_xor(rs, 2, 32);
  if (quarter == 0) asum_part[((size_t)n * 16 + pp) * KC + krow] = rs;
}

// K3b: reduce 16 panel partials -> asum[n,k] (deterministic, no atomics).
__global__ __launch_bounds__(256) void k3b_reduce(const float* __restrict__ part,
                                                  float* __restrict__ asum) {
  const int i = blockIdx.x * 256 + threadIdx.x;
  if (i >= NB * KC) return;
  const int n = i >> 6, k = i & 63;
  float s = 0.f;
#pragma unroll
  for (int pnl = 0; pnl < 16; ++pnl) s += part[((size_t)n * 16 + pnl) * KC + k];
  asum[i] = s;
}

// ---------------------------------------------------------------------------
// K4: vlad[k,d] = sum_p a[k,p]*xf[p,d] - c[k,d]*asum[k]  (WMMA over p=1024),
// then intra-normalize rows and global normalize; one block per image.
// ---------------------------------------------------------------------------
__global__ __launch_bounds__(256) void k4_vlad(const _Float16* __restrict__ xf16,
                                               const _Float16* __restrict__ a_in,
                                               const float* __restrict__ asum,
                                               const float* __restrict__ centroids,
                                               float* __restrict__ out) {
  __shared__ __align__(16) char smem[53248];
  _Float16* smA = (_Float16*)smem;               // a panel [64k][p] 17408 B
  _Float16* smY = (_Float16*)(smem + 17408);     // xf panel [128d][p] 34816 B
  float* vbuf   = (float*)smem;                  // reuse: [64][132] f32 33792 B
  float* rowsc  = (float*)(smem + 33792);        // 64 row scales
  float* rowss  = (float*)(smem + 34048);        // 64 row sumsq (post-norm)
  const int n = blockIdx.x;
  const int tid = threadIdx.x, wave = tid >> 5, lane = tid & 31;
  const int kt = wave >> 1;                      // k row tile 0..3
  const int cg = (wave & 1) * 4;                 // d col tile group base

  const v8f zero8 = {0.f, 0.f, 0.f, 0.f, 0.f, 0.f, 0.f, 0.f};
  v8f c[4] = {zero8, zero8, zero8, zero8};

  for (int pp = 0; pp < 8; ++pp) {               // 8 K-panels of 128 pixels
    __syncthreads();
    stage_rows(smA, a_in + (size_t)n * KC * PN + pp * 128, 64, tid);
    stage_rows(smY, xf16 + (size_t)n * DD * PN + pp * 128, 128, tid);
    __syncthreads();
#pragma unroll
    for (int kc = 0; kc < 4; ++kc) {
      v16h af = load_a_frag(smA, kt * 16, kc * 32, lane);
      v16h bb[4];
#pragma unroll
      for (int ct = 0; ct < 4; ++ct)
        bb[ct] = load_b_frag(smY, (cg + ct) * 16, kc * 32, lane);
#pragma unroll
      for (int ct = 0; ct < 4; ++ct)
        c[ct] = __builtin_amdgcn_wmma_f32_16x16x32_f16(false, af, false, bb[ct],
                                                       (short)0, c[ct], false, false);
    }
  }
  __syncthreads();  // matmul done; smem is now reused as vbuf

#pragma unroll
  for (int ct = 0; ct < 4; ++ct) {
    const int d = (cg + ct) * 16 + (lane & 15);
#pragma unroll
    for (int r = 0; r < 8; ++r) {
      const int k = kt * 16 + r + 8 * (lane >> 4);
      vbuf[k * 132 + d] = c[ct][r] - centroids[k * DD + d] * asum[n * KC + k];
    }
  }
  __syncthreads();

  // Intra (per-cluster) normalization: 4 threads per row of 128.
  const int row = tid >> 2, part = tid & 3;
  float ss = 0.f;
#pragma unroll
  for (int i = 0; i < 32; ++i) {
    const float v = vbuf[row * 132 + part * 32 + i];
    ss = fmaf(v, v, ss);
  }
  ss += __shfl_xor(ss, 1, 32);
  ss += __shfl_xor(ss, 2, 32);
  const float sc = 1.0f / fmaxf(sqrtf(ss), 1e-12f);
  if (part == 0) { rowsc[row] = sc; rowss[row] = ss * sc * sc; }
  __syncthreads();

  // Global normalization (deterministic: every thread sums the same 64 values).
  float tot = 0.f;
#pragma unroll
  for (int r = 0; r < KC; ++r) tot += rowss[r];
  const float gsc = 1.0f / fmaxf(sqrtf(tot), 1e-12f);

  for (int it = 0; it < 32; ++it) {
    const int idx = it * 256 + tid;
    const int k = idx >> 7, d = idx & 127;
    out[(size_t)n * (KC * DD) + idx] = vbuf[k * 132 + d] * rowsc[k] * gsc;
  }
}

// ---------------------------------------------------------------------------
// Workspace layout (bytes):
//   [0,          8388608)  xf16   f16 [N,D,P]
//   [8388608,    8519680)  invw   f32 [N,P]
//   [8519680,   12713984)  a      f16 [N,K,P]
//   [12713984,  12845056)  asum_part f32 [N,16,K]
//   [12845056,  12853248)  asum   f32 [N,K]
// ---------------------------------------------------------------------------
extern "C" void kernel_launch(void* const* d_in, const int* in_sizes, int n_in,
                              void* d_out, int out_size, void* d_ws, size_t ws_size,
                              hipStream_t stream) {
  (void)in_sizes; (void)n_in; (void)out_size; (void)ws_size;
  const float* x         = (const float*)d_in[0];
  const float* conv_w    = (const float*)d_in[1];
  const float* centroids = (const float*)d_in[2];
  const float* abp       = (const float*)d_in[3];
  float* out = (float*)d_out;

  char* ws = (char*)d_ws;
  _Float16* xf16     = (_Float16*)(ws);
  float*    invw     = (float*)(ws + 8388608);
  _Float16* a_ws     = (_Float16*)(ws + 8519680);
  float*    asum_prt = (float*)(ws + 12713984);
  float*    asum     = (float*)(ws + 12845056);

  k1_norm     <<<dim3(4, NB),  256, 0, stream>>>(x, xf16);
  k2_antiburst<<<dim3(8, NB),  256, 0, stream>>>(xf16, abp, invw);
  k3_assign   <<<dim3(16, NB), 256, 0, stream>>>(xf16, conv_w, invw, a_ws, asum_prt);
  k3b_reduce  <<<8,            256, 0, stream>>>(asum_prt, asum);
  k4_vlad     <<<NB,           256, 0, stream>>>(xf16, a_ws, asum, centroids, out);
}
